// BSDE_solver_43250320671520
// MI455X (gfx1250) — matrix operand order repaired and checked by reading, not compiled, not run on hardware
//
#include <hip/hip_runtime.h>
#include <hip/hip_bf16.h>

typedef __attribute__((ext_vector_type(16))) _Float16 v16h;
typedef __attribute__((ext_vector_type(8)))  float    v8f;
typedef __attribute__((ext_vector_type(4)))  unsigned int v4u;
typedef __attribute__((ext_vector_type(8)))  int      v8i;
typedef __attribute__((ext_vector_type(4)))  int      v4i;

#define DEV __device__ __forceinline__

// Tensor Data Mover path (this toolchain: 6-arg builtin, per round-2 diagnostic)
#if defined(__gfx1250__) && __has_builtin(__builtin_amdgcn_tensor_load_to_lds) && \
    __has_builtin(__builtin_amdgcn_s_wait_tensorcnt)
#define USE_TDM 1
#else
#define USE_TDM 0
#endif

// ---------------- problem constants ----------------
constexpr int B_   = 8192;   // batch (paths)
constexpr int DIM_ = 100;    // assets
constexpr int NT_  = 50;     // timesteps
constexpr int W_   = 256;    // MLP width
constexpr float R_ = 0.05f;  // risk-free rate

constexpr int BM   = 64;     // batch rows per workgroup
constexpr int NTHR = 128;    // 4 wave32 waves per workgroup

// ---------------- f16 weight workspace layout (units: halfs) ----------------
// All weights stored TRANSPOSED: Wt[n][k] = W[k][n], K padded for 32-chunks.
constexpr int SZ_IN    = 256 * 128;          // [256][128] (k<101 valid)
constexpr int SZ_H     = 3 * 256 * 256;      // [3][256][256]
constexpr int SZ_GOUT  = 112 * 256;          // [112][256] (n<100 valid)
constexpr int SZ_VOUT  = 16 * 256;           // [16][256]  (n<1 valid)
constexpr int SZ_VT    = 112 * 128;          // V itself: Vt[n][k]=V[n][k]
constexpr int OFF_GIN  = 0;
constexpr int OFF_GH   = OFF_GIN + SZ_IN;
constexpr int OFF_GOUT = OFF_GH + SZ_H;
constexpr int OFF_VIN  = OFF_GOUT + SZ_GOUT;
constexpr int OFF_VH   = OFF_VIN + SZ_IN;
constexpr int OFF_VOUT = OFF_VH + SZ_H;
constexpr int OFF_VT   = OFF_VOUT + SZ_VOUT;
constexpr int WS_HALFS = OFF_VT + SZ_VT;     // ~506K halfs ~= 1.0 MB

// ---------------- LDS layout ----------------
constexpr int L_INPV  = 0;                   // [64][128] f16 MLP input [t, S]
constexpr int L_A0    = L_INPV + BM * 128;   // [64][256] f16 activations
constexpr int L_A1    = L_A0   + BM * 256;   // [64][256] f16 activations
constexpr int L_WBUF  = L_A1   + BM * 256;   // [256][256] f16 layer weights
constexpr int L_VT    = L_WBUF + 256 * 256;  // [112][128] f16 resident V
constexpr int L_DWB   = L_VT   + SZ_VT;      // [64][128] f16 dW tile
constexpr int L_HALFS = L_DWB  + BM * 128;
constexpr int LF_S    = 0;                   // [64][112] f32 state S
constexpr int LF_VOLD = LF_S + BM * 112;     // [64] f32 v_old
constexpr int LF_STO  = LF_VOLD + BM;        // [64] f32 stoch integral
constexpr int LF_ERR  = LF_STO + BM;         // [64] f32 error accumulator
constexpr int LF_CNT  = LF_ERR + BM;
constexpr size_t SMEM_BYTES = (size_t)L_HALFS * 2 + (size_t)LF_CNT * 4;
static_assert(SMEM_BYTES <= 320u * 1024u, "exceeds WGP LDS");

// ---------------- WMMA fragment helpers (wave32, 16x16x32 f16) ----------------
// A 16x32 (MxK): lane L -> M=L&15; lanes<16 hold K {0..7,16..23}, lanes>=16 {8..15,24..31}
DEV v16h ld_a(const _Float16* base, int row, int stride, int k0, int lane) {
  const _Float16* q = base + row * stride + k0 + ((lane & 16) ? 8 : 0);
  v16h r;
  reinterpret_cast<uint4*>(&r)[0] = *reinterpret_cast<const uint4*>(q);
  reinterpret_cast<uint4*>(&r)[1] = *reinterpret_cast<const uint4*>(q + 16);
  return r;
}
// B 32x16 (KxN) from column-major Wt[n][k]: lane L -> N=L&15; lanes<16 K {0..7,16..23}, lanes>=16 {8..15,24..31}
DEV v16h ld_b(const _Float16* wt, int n, int stride, int k0, int lane) {
  const _Float16* q = wt + n * stride + k0 + ((lane & 16) ? 8 : 0);
  v16h r;
  reinterpret_cast<uint4*>(&r)[0] = *reinterpret_cast<const uint4*>(q);
  reinterpret_cast<uint4*>(&r)[1] = *reinterpret_cast<const uint4*>(q + 16);
  return r;
}
DEV v8f wmma16(v16h a, v16h b, v8f c) {
  return __builtin_amdgcn_wmma_f32_16x16x32_f16(false, a, false, b, (short)0, c, false, false);
}

// ---- global(f16) -> LDS panel staging ----
// Preferred: one TDM descriptor per panel (TENSORcnt, no VGPR traffic).
// Fallback: cooperative b128 copy through VGPRs.
DEV void stage(const _Float16* g, _Float16* l, int halfs, int tid) {
#if USE_TDM
  if (tid < 32) {  // wave 0 issues the DMA for the workgroup
    unsigned long long ga = (unsigned long long)(const void*)g;
    unsigned lo = (unsigned)(unsigned long long)(void*)l;  // LDS aperture: low 32b = LDS offset
    unsigned e  = (unsigned)halfs >> 2;                    // 8-byte elements (<= 16384)
    // D# group0: count=1 | lds_addr | global_addr[56:0] | type=2 ("image")
    v4u g0 = { 1u, lo, (unsigned)ga, (unsigned)((ga >> 32) & 0x1FFFFFFu) | (2u << 30) };
    // D# group1: data_size=3 (8B); tensor_dim0=e; tensor_dim1=1; tile_dim0=e;
    //            tile_dim1=1; tensor_dim0_stride=e; no pad/iterate/multicast.
    v8i g1 = { (int)(3u << 16),
               (int)((e & 0xFFFFu) << 16),
               (int)((e >> 16) | (1u << 16)),
               (int)((e & 0xFFFFu) << 16),
               1,
               (int)e,
               0,
               0 };
    v4i z4 = { 0, 0, 0, 0 };
    v8i z8 = { 0, 0, 0, 0, 0, 0, 0, 0 };
    __builtin_amdgcn_tensor_load_to_lds(g0, g1, z4, z4, z8, 0);
    __builtin_amdgcn_s_wait_tensorcnt(0);
  }
#else
  const uint4* gs = reinterpret_cast<const uint4*>(g);
  uint4* ls = reinterpret_cast<uint4*>(l);
  int vecs = halfs >> 3;
  for (int i = tid; i < vecs; i += NTHR) ls[i] = gs[i];
#endif
}

// one dense layer: out[64x256] = relu(in[64xK] @ Wt + bias). wt stride == istride.
template <int KT>
DEV void layer(const _Float16* in, int istride, const _Float16* wt,
               const float* __restrict__ bias, _Float16* out, int wave, int lane) {
  const int row0 = wave * 16;
  const int nl   = lane & 15;
  const int hoff = (lane & 16) ? 8 : 0;
  const int arow = row0 + nl;
  for (int nt = 0; nt < 16; ++nt) {
    const int n = nt * 16 + nl;
    v8f acc = {};
#pragma unroll
    for (int kt = 0; kt < KT; ++kt) {
      v16h a = ld_a(in, arow, istride, kt * 32, lane);
      v16h b = ld_b(wt, n, istride, kt * 32, lane);
      acc = wmma16(a, b, acc);
    }
    const float bv = bias[n];
#pragma unroll
    for (int g = 0; g < 8; ++g) {
      float x = acc[g] + bv;
      x = fmaxf(x, 0.f);
      out[(row0 + g + hoff) * 256 + n] = (_Float16)x;
    }
  }
}

// input layer + 3 hidden layers; leaves final-layer input in a0, final Wt staged in wbuf
DEV void trunk(const _Float16* ws, int off_in, int off_h, int off_out, int out_halfs,
               const float* b_in, const float* b_h,
               _Float16* inpv, _Float16* a0, _Float16* a1, _Float16* wbuf,
               int tid, int wave, int lane) {
  stage(ws + off_in, wbuf, SZ_IN, tid);               __syncthreads();
  layer<4>(inpv, 128, wbuf, b_in, a1, wave, lane);    __syncthreads();
  stage(ws + off_h, wbuf, 65536, tid);                __syncthreads();
  layer<8>(a1, 256, wbuf, b_h, a0, wave, lane);       __syncthreads();
  stage(ws + off_h + 65536, wbuf, 65536, tid);        __syncthreads();
  layer<8>(a0, 256, wbuf, b_h + 256, a1, wave, lane); __syncthreads();
  stage(ws + off_h + 131072, wbuf, 65536, tid);       __syncthreads();
  layer<8>(a1, 256, wbuf, b_h + 512, a0, wave, lane); __syncthreads();
  stage(ws + off_out, wbuf, out_halfs, tid);          __syncthreads();
}

// value-net head: vnew = a0 @ Wt_out + b; BSDE error update
DEV void value_final(const _Float16* a0, const _Float16* wbuf, const float* bv_out,
                     float* vOld, const float* stoch, float* errA,
                     float rh, bool first, int wave, int lane) {
  const int row0 = wave * 16;
  const int nl   = lane & 15;
  const int hoff = (lane & 16) ? 8 : 0;
  v8f acc = {};
#pragma unroll
  for (int kt = 0; kt < 8; ++kt) {
    v16h a = ld_a(a0, row0 + nl, 256, kt * 32, lane);
    v16h b = ld_b(wbuf, nl, 256, kt * 32, lane);
    acc = wmma16(a, b, acc);
  }
  if (nl == 0) {
    const float bv = bv_out[0];
#pragma unroll
    for (int g = 0; g < 8; ++g) {
      const int row = row0 + g + hoff;
      const float vnew = acc[g] + bv;
      if (!first) {
        const float e = vnew - vOld[row] * (1.f + rh) - stoch[row];
        errA[row] += e * e;
      }
      vOld[row] = vnew;
    }
  }
}

// grad-net head fused with vol = S*sqrt(h)*(dW @ V^T), stoch integral, S update,
// and next value-net input [.,S_new] refresh.
DEV void grad_final_vol(const _Float16* a0, const _Float16* wbuf, const float* bg_out,
                        const _Float16* dwb, const _Float16* vt,
                        float* Sl, float* stoch, _Float16* inpv,
                        float rh, float sdt, int wave, int lane) {
  const int row0 = wave * 16;
  const int nl   = lane & 15;
  const int hoff = (lane & 16) ? 8 : 0;
  for (int ntile = 0; ntile < 7; ++ntile) {   // 112 padded output cols
    const int n = ntile * 16 + nl;
    v8f gacc = {};
    v8f vacc = {};
#pragma unroll
    for (int kt = 0; kt < 8; ++kt) {          // grad = a0 @ gWt_out
      v16h a = ld_a(a0, row0 + nl, 256, kt * 32, lane);
      v16h b = ld_b(wbuf, n, 256, kt * 32, lane);
      gacc = wmma16(a, b, gacc);
    }
#pragma unroll
    for (int kt = 0; kt < 4; ++kt) {          // dvol = dW @ V^T
      v16h a = ld_a(dwb, row0 + nl, 128, kt * 32, lane);
      v16h b = ld_b(vt, n, 128, kt * 32, lane);
      vacc = wmma16(a, b, vacc);
    }
    if (n < DIM_) {
      const float bg = bg_out[n];
#pragma unroll
      for (int g = 0; g < 8; ++g) {
        const int row = row0 + g + hoff;
        const float sold = Sl[row * 112 + n];
        const float vol  = sold * sdt * vacc[g];
        const float grad = gacc[g] + bg;
        atomicAdd(&stoch[row], grad * vol);           // ds_add_f32 row-reduction
        const float snew = sold * (1.f + rh) + vol;
        Sl[row * 112 + n] = snew;
        inpv[row * 128 + n + 1] = (_Float16)snew;     // next MLP input S part
      }
    }
  }
}

// ---------------- prologue: fp32 weights -> f16 transposed/padded workspace ----------------
__global__ void repack_weights(const float* __restrict__ Wg_in, const float* __restrict__ Wg_h,
                               const float* __restrict__ Wg_out, const float* __restrict__ Wv_in,
                               const float* __restrict__ Wv_h, const float* __restrict__ Wv_out,
                               const float* __restrict__ V, _Float16* __restrict__ ws) {
  const int tid = blockIdx.x * blockDim.x + threadIdx.x;
  const int np  = gridDim.x * blockDim.x;
  for (int i = tid; i < SZ_IN; i += np) {            // [256][128] <- [101][256]
    int n = i >> 7, k = i & 127;
    ws[OFF_GIN + i] = (_Float16)((k < DIM_ + 1) ? Wg_in[k * W_ + n] : 0.f);
    ws[OFF_VIN + i] = (_Float16)((k < DIM_ + 1) ? Wv_in[k * W_ + n] : 0.f);
  }
  for (int i = tid; i < SZ_H; i += np) {             // [3][256][256] transposed
    int l = i >> 16, rem = i & 65535;
    int n = rem >> 8, k = rem & 255;
    size_t src = ((size_t)l * W_ + k) * W_ + n;
    ws[OFF_GH + i] = (_Float16)Wg_h[src];
    ws[OFF_VH + i] = (_Float16)Wv_h[src];
  }
  for (int i = tid; i < SZ_GOUT; i += np) {          // [112][256] <- [256][100]
    int n = i >> 8, k = i & 255;
    ws[OFF_GOUT + i] = (_Float16)((n < DIM_) ? Wg_out[k * DIM_ + n] : 0.f);
  }
  for (int i = tid; i < SZ_VOUT; i += np) {          // [16][256] <- [256][1]
    int n = i >> 8, k = i & 255;
    ws[OFF_VOUT + i] = (_Float16)((n == 0) ? Wv_out[k] : 0.f);
  }
  for (int i = tid; i < SZ_VT; i += np) {            // [112][128] = V (B of dW@V^T)
    int n = i >> 7, k = i & 127;
    ws[OFF_VT + i] = (_Float16)((n < DIM_ && k < DIM_) ? V[n * DIM_ + k] : 0.f);
  }
}

// ---------------- fused persistent BSDE scan ----------------
__global__ __launch_bounds__(NTHR) void bsde_main(
    const float* __restrict__ S0, const float* __restrict__ dW,
    const float* __restrict__ tg, const _Float16* __restrict__ ws,
    const float* __restrict__ bg_in, const float* __restrict__ bg_h,
    const float* __restrict__ bg_out, const float* __restrict__ bv_in,
    const float* __restrict__ bv_h, const float* __restrict__ bv_out,
    float* __restrict__ out) {
  extern __shared__ char smem[];
  _Float16* hs   = (_Float16*)smem;
  _Float16* inpv = hs + L_INPV;
  _Float16* a0   = hs + L_A0;
  _Float16* a1   = hs + L_A1;
  _Float16* wbuf = hs + L_WBUF;
  _Float16* vt   = hs + L_VT;
  _Float16* dwb  = hs + L_DWB;
  float* fs    = (float*)(smem + (size_t)L_HALFS * 2);
  float* Sl    = fs + LF_S;
  float* vOld  = fs + LF_VOLD;
  float* stoch = fs + LF_STO;
  float* errA  = fs + LF_ERR;

  const int tid   = threadIdx.x;
  const int wave  = tid >> 5;
  const int lane  = tid & 31;
  const int rbase = blockIdx.x * BM;

  stage(ws + OFF_VT, vt, SZ_VT, tid);   // resident V tile

  // build [t0, S0] input and fp32 state
  const float t0 = tg[0];
  for (int i = tid; i < BM * 128; i += NTHR) {
    int row = i >> 7, k = i & 127;
    float val = 0.f;
    if (k == 0) val = t0;
    else if (k <= DIM_) {
      float s = S0[(size_t)(rbase + row) * DIM_ + (k - 1)];
      Sl[row * 112 + (k - 1)] = s;
      val = s;
    }
    inpv[i] = (_Float16)val;
  }
  if (tid < BM) { errA[tid] = 0.f; stoch[tid] = 0.f; }
  __syncthreads();

  // v0 = valuenet([t0, S0])
  trunk(ws, OFF_VIN, OFF_VH, OFF_VOUT, SZ_VOUT, bv_in, bv_h, inpv, a0, a1, wbuf, tid, wave, lane);
  value_final(a0, wbuf, bv_out, vOld, stoch, errA, 0.f, true, wave, lane);
  __syncthreads();

#pragma unroll 1
  for (int t = 0; t < NT_ - 1; ++t) {
    const float tp = tg[t], tc = tg[t + 1];
    const float h   = tc - tp;
    const float rh  = R_ * h;
    const float sdt = sqrtf(h);

    // dW tile -> f16 LDS (the only streamed HBM traffic: 160 MB total)
    const float* dWt = dW + ((size_t)t * B_ + rbase) * DIM_;
    for (int i = tid; i < BM * 128; i += NTHR) {
      int row = i >> 7, k = i & 127;
      dwb[i] = (_Float16)((k < DIM_) ? dWt[(size_t)row * DIM_ + k] : 0.f);
    }
    if (tid < BM) stoch[tid] = 0.f;

    // prefetch next step's dW tile into cache while both MLP trunks run
    if (t + 1 < NT_ - 1) {
      const char* nxt = (const char*)(dW + ((size_t)(t + 1) * B_ + rbase) * DIM_);
      for (int off = tid * 256; off < BM * DIM_ * 4; off += NTHR * 256)
        __builtin_prefetch(nxt + off, 0, 3);          // global_prefetch_b8
    }

    // grad = gradnet([tp, S_old]); fused head computes vol, stoch, S_new, next input
    trunk(ws, OFF_GIN, OFF_GH, OFF_GOUT, SZ_GOUT, bg_in, bg_h, inpv, a0, a1, wbuf, tid, wave, lane);
    grad_final_vol(a0, wbuf, bg_out, dwb, vt, Sl, stoch, inpv, rh, sdt, wave, lane);
    __syncthreads();
    if (tid < BM) inpv[tid * 128] = (_Float16)tc;   // time column

    // vNew = valuenet([tc, S_new]); error update
    trunk(ws, OFF_VIN, OFF_VH, OFF_VOUT, SZ_VOUT, bv_in, bv_h, inpv, a0, a1, wbuf, tid, wave, lane);
    value_final(a0, wbuf, bv_out, vOld, stoch, errA, rh, false, wave, lane);
    __syncthreads();
  }

  // outputs: v_f [B], S_f [B,DIM], error [B] concatenated
  if (tid < BM) {
    out[rbase + tid] = vOld[tid];
    out[(size_t)B_ + (size_t)B_ * DIM_ + rbase + tid] = errA[tid];
  }
  for (int i = tid; i < BM * DIM_; i += NTHR) {
    int row = i / DIM_, k = i % DIM_;
    out[(size_t)B_ + (size_t)(rbase + row) * DIM_ + k] = Sl[row * 112 + k];
  }
}

extern "C" void kernel_launch(void* const* d_in, const int* in_sizes, int n_in,
                              void* d_out, int out_size, void* d_ws, size_t ws_size,
                              hipStream_t stream) {
  (void)in_sizes; (void)n_in; (void)out_size; (void)ws_size;
  const float* S0     = (const float*)d_in[0];
  const float* dW     = (const float*)d_in[1];
  const float* tg     = (const float*)d_in[2];
  const float* V      = (const float*)d_in[3];
  const float* Wg_in  = (const float*)d_in[4];
  const float* bg_in  = (const float*)d_in[5];
  const float* Wg_h   = (const float*)d_in[6];
  const float* bg_h   = (const float*)d_in[7];
  const float* Wg_out = (const float*)d_in[8];
  const float* bg_out = (const float*)d_in[9];
  const float* Wv_in  = (const float*)d_in[10];
  const float* bv_in  = (const float*)d_in[11];
  const float* Wv_h   = (const float*)d_in[12];
  const float* bv_h   = (const float*)d_in[13];
  const float* Wv_out = (const float*)d_in[14];
  const float* bv_out = (const float*)d_in[15];
  _Float16* ws = (_Float16*)d_ws;
  float* out = (float*)d_out;

  repack_weights<<<dim3(256), dim3(256), 0, stream>>>(Wg_in, Wg_h, Wg_out, Wv_in, Wv_h,
                                                      Wv_out, V, ws);

  (void)hipFuncSetAttribute((const void*)bsde_main,
                            hipFuncAttributeMaxDynamicSharedMemorySize, (int)SMEM_BYTES);
  bsde_main<<<dim3(B_ / BM), dim3(NTHR), SMEM_BYTES, stream>>>(
      S0, dW, tg, ws, bg_in, bg_h, bg_out, bv_in, bv_h, bv_out, out);
}
